// CFGDualCrossAttention_77412490543667
// MI455X (gfx1250) — matrix operand order repaired and checked by reading, not compiled, not run on hardware
//
#include <hip/hip_runtime.h>
#include <hip/hip_bf16.h>

#define DIMX   5120
#define NHEADS 40
#define HDIM   128
#define SEQ    4096
#define CTX    512
#define CBATCH 2

typedef __bf16 bf16_t;
typedef bf16_t v16bf __attribute__((ext_vector_type(16)));
typedef float  v8f   __attribute__((ext_vector_type(8)));

union Frag16 { v16bf v; unsigned int u[8]; };

__device__ __forceinline__ unsigned int bf16pack2(float lo, float hi) {
  unsigned int a = __float_as_uint(lo);
  unsigned int b = __float_as_uint(hi);
  a = (a + 0x7FFFu + ((a >> 16) & 1u)) >> 16;
  b = (b + 0x7FFFu + ((b >> 16) & 1u)) >> 16;
  return (a & 0xFFFFu) | (b << 16);
}
__device__ __forceinline__ unsigned short bf16one(float f) {
  unsigned int x = __float_as_uint(f);
  x = (x + 0x7FFFu + ((x >> 16) & 1u)) >> 16;
  return (unsigned short)x;
}

// ---------------------------------------------------------------------------
// Tiled WMMA GEMM:  C[M,N] = A[M,K] * W[N,K]^T + bias,  fp32 accumulate.
// A is fp32 (projections) or bf16 (attention output); W/bias fp32; C fp32.
// Block = 256 threads (8 waves), tile 128x128, BK=32. Each wave owns 32x64.
//
// Grid order: blockIdx.x = M-tile (innermost). Consecutive blocks then share
// one 128-col weight strip (2.6 MB) and re-stream the A matrix, whose whole
// working set (<= 84 MB) fits the 192 MB L2. This guarantees each weight byte
// is fetched from HBM exactly once — critical for Wkv (210 MB > L2), which
// would otherwise be re-streamed per M-row.
// ---------------------------------------------------------------------------
template <bool ABF16>
__global__ __launch_bounds__(256)
void gemm_wmma_bf16(const void* __restrict__ Ap, const float* __restrict__ W,
                    const float* __restrict__ bias, float* __restrict__ C,
                    int M, int N, int K) {
  __shared__ unsigned int As[128 * 17];  // 128 rows x 32 bf16 (16 uints) + 1 pad
  __shared__ unsigned int Bs[128 * 17];

  const int tid  = threadIdx.x;
  const int lane = tid & 31;
  const int wave = tid >> 5;
  const int hl   = lane >> 4;   // which 16-lane half
  const int lidx = lane & 15;
  const int m0 = blockIdx.x * 128;   // M innermost (see note above)
  const int n0 = blockIdx.y * 128;
  const int wm = (wave >> 1) * 32;  // wave row offset in tile
  const int wn = (wave & 1) * 64;   // wave col offset in tile

  const v8f zero8 = {0.f, 0.f, 0.f, 0.f, 0.f, 0.f, 0.f, 0.f};
  v8f acc[2][4];
#pragma unroll
  for (int i = 0; i < 2; i++)
#pragma unroll
    for (int j = 0; j < 4; j++) acc[i][j] = zero8;

  const int lr   = tid >> 1;        // 0..127 staging row
  const int lseg = (tid & 1) * 16;  // k offset 0 / 16
  const float* Af          = (const float*)Ap;
  const unsigned short* Ab = (const unsigned short*)Ap;

  for (int kt = 0; kt < K; kt += 32) {
    // ---- stage A tile (convert fp32 -> packed bf16 pairs) ----
    unsigned int pk[8];
    if (ABF16) {
      const uint4* p = (const uint4*)(Ab + (size_t)(m0 + lr) * K + kt + lseg);
      uint4 x0 = p[0], x1 = p[1];
      pk[0] = x0.x; pk[1] = x0.y; pk[2] = x0.z; pk[3] = x0.w;
      pk[4] = x1.x; pk[5] = x1.y; pk[6] = x1.z; pk[7] = x1.w;
    } else {
      const float4* p = (const float4*)(Af + (size_t)(m0 + lr) * K + kt + lseg);
      float4 f0 = p[0], f1 = p[1], f2 = p[2], f3 = p[3];
      pk[0] = bf16pack2(f0.x, f0.y); pk[1] = bf16pack2(f0.z, f0.w);
      pk[2] = bf16pack2(f1.x, f1.y); pk[3] = bf16pack2(f1.z, f1.w);
      pk[4] = bf16pack2(f2.x, f2.y); pk[5] = bf16pack2(f2.z, f2.w);
      pk[6] = bf16pack2(f3.x, f3.y); pk[7] = bf16pack2(f3.z, f3.w);
    }
    {
      unsigned int* d = &As[lr * 17 + (lseg >> 1)];
#pragma unroll
      for (int i = 0; i < 8; i++) d[i] = pk[i];
    }
    // ---- stage W tile (rows = output col n, K-contiguous) ----
    {
      const float4* p = (const float4*)(W + (size_t)(n0 + lr) * K + kt + lseg);
      float4 f0 = p[0], f1 = p[1], f2 = p[2], f3 = p[3];
      unsigned int* d = &Bs[lr * 17 + (lseg >> 1)];
      d[0] = bf16pack2(f0.x, f0.y); d[1] = bf16pack2(f0.z, f0.w);
      d[2] = bf16pack2(f1.x, f1.y); d[3] = bf16pack2(f1.z, f1.w);
      d[4] = bf16pack2(f2.x, f2.y); d[5] = bf16pack2(f2.z, f2.w);
      d[6] = bf16pack2(f3.x, f3.y); d[7] = bf16pack2(f3.z, f3.w);
    }
    __syncthreads();

    // ---- fragments + WMMA ----
    Frag16 bfr[4];
#pragma unroll
    for (int fn = 0; fn < 4; fn++) {
      const int bcol = wn + fn * 16 + lidx;  // B: lane holds col, VGPR holds K pair
#pragma unroll
      for (int v = 0; v < 8; v++) bfr[fn].u[v] = Bs[bcol * 17 + hl * 8 + v];
    }
#pragma unroll
    for (int fm = 0; fm < 2; fm++) {
      Frag16 a;
      const int arow = wm + fm * 16 + lidx;  // A: lane holds row
#pragma unroll
      for (int v = 0; v < 8; v++)
        a.u[v] = As[arow * 17 + hl * 8 + v];
#pragma unroll
      for (int fn = 0; fn < 4; fn++)
        acc[fm][fn] = __builtin_amdgcn_wmma_f32_16x16x32_bf16(
            false, a.v, false, bfr[fn].v, (short)0, acc[fm][fn], false, false);
    }
    __syncthreads();
  }

  // ---- epilogue: bias + fp32 store ----
#pragma unroll
  for (int fm = 0; fm < 2; fm++)
#pragma unroll
    for (int fn = 0; fn < 4; fn++) {
      const int gn = n0 + wn + fn * 16 + lidx;
      const float bv = bias[gn];
#pragma unroll
      for (int r = 0; r < 8; r++) {
        const int gm = m0 + wm + fm * 16 + hl * 8 + r;
        C[(size_t)gm * N + gn] = acc[fm][fn][r] + bv;
      }
    }
}

// ---------------------------------------------------------------------------
// Row RMSNorm (fp32) -> bf16 out; optional extra scale (folds Dh^-0.5 into Q).
// One block per row, 256 threads, D = 5120.
// ---------------------------------------------------------------------------
__global__ __launch_bounds__(256)
void rmsnorm_to_bf16(const float* __restrict__ X, long ldx,
                     const float* __restrict__ g,
                     unsigned short* __restrict__ out, float extra_scale) {
  const int row = blockIdx.x;
  const int t   = threadIdx.x;
  const float* x = X + (size_t)row * ldx;
  float vals[DIMX / 256];
  float ss = 0.f;
#pragma unroll
  for (int i = 0; i < DIMX / 256; i++) {
    float v = x[t + i * 256];
    vals[i] = v;
    ss += v * v;
  }
  __shared__ float red[256];
  red[t] = ss;
  __syncthreads();
#pragma unroll
  for (int off = 128; off >= 1; off >>= 1) {
    if (t < off) red[t] += red[t + off];
    __syncthreads();
  }
  const float inv = rsqrtf(red[0] * (1.0f / DIMX) + 1e-6f) * extra_scale;
  unsigned short* o = out + (size_t)row * DIMX;
#pragma unroll
  for (int i = 0; i < DIMX / 256; i++) {
    const int c = t + i * 256;
    o[c] = bf16one(vals[i] * inv * g[c]);
  }
}

// ---------------------------------------------------------------------------
// V slice of kv [CB*L, 2D] -> per-head transposed bf16 vT [CB,H,Dh,L]
// (so PV B-fragments become contiguous dword loads). 32L x 128D tile via LDS.
// ---------------------------------------------------------------------------
__global__ __launch_bounds__(256)
void vtrans_kernel(const float* __restrict__ kv, unsigned short* __restrict__ vT) {
  const int cb = blockIdx.x >> 4;          // CTX/32 = 16 blocks per cb
  const int l0 = (blockIdx.x & 15) * 32;
  const int h  = blockIdx.y;
  __shared__ unsigned short tile[32][130];
  const int t = threadIdx.x;
  {
    const int lr = t >> 3;
    const int c0 = (t & 7) * 16;
    const float4* src = (const float4*)(kv + (size_t)(cb * CTX + l0 + lr) * (2 * DIMX) +
                                        DIMX + h * HDIM + c0);
#pragma unroll
    for (int i = 0; i < 4; i++) {
      float4 f = src[i];
      tile[lr][c0 + i * 4 + 0] = bf16one(f.x);
      tile[lr][c0 + i * 4 + 1] = bf16one(f.y);
      tile[lr][c0 + i * 4 + 2] = bf16one(f.z);
      tile[lr][c0 + i * 4 + 3] = bf16one(f.w);
    }
  }
  __syncthreads();
  {
    const int d   = t >> 1;
    const int lc0 = (t & 1) * 16;
    unsigned short* dst =
        vT + ((size_t)(cb * NHEADS + h) * HDIM + d) * CTX + l0 + lc0;
#pragma unroll
    for (int i = 0; i < 16; i++) dst[i] = tile[lc0 + i][d];
  }
}

// ---------------------------------------------------------------------------
// Flash attention: one wave per (cb, head, 16 query rows). L-blocks of 32,
// QK^T + online softmax + PV, all through v_wmma_f32_16x16x32_bf16.
// Scale (Dh^-0.5) is pre-folded into qn.
// Row-sum reduction across lanes is DEFERRED to after the L loop: all 16
// lanes of a half share m_new/corr, so per-lane partial sums scale
// identically — this halves the ds_bpermute shuffle traffic per L-block.
// ---------------------------------------------------------------------------
__global__ __launch_bounds__(32)
void attn_kernel(const unsigned short* __restrict__ qn,
                 const unsigned short* __restrict__ kn,
                 const unsigned short* __restrict__ vT,
                 unsigned short* __restrict__ out) {
  const int s0   = blockIdx.x * 16;
  const int h    = blockIdx.y;
  const int cb   = blockIdx.z;
  const int lane = threadIdx.x;
  const int hl   = lane >> 4;
  const int lidx = lane & 15;
  __shared__ unsigned short Pt[16 * 34];  // 16 x 32 bf16 P tile (+pad)

  // Q A-fragments: 16 rows x 128 (4 frags of K=32)
  Frag16 qa[4];
  {
    const unsigned short* qrow =
        qn + (size_t)(cb * SEQ + s0 + lidx) * DIMX + h * HDIM;
#pragma unroll
    for (int f = 0; f < 4; f++)
#pragma unroll
      for (int v = 0; v < 8; v++) {
        const int k0 = f * 32 + hl * 16 + v * 2;
        qa[f].u[v] = *(const unsigned int*)(qrow + k0);
      }
  }

  const v8f zero8 = {0.f, 0.f, 0.f, 0.f, 0.f, 0.f, 0.f, 0.f};
  v8f O[8];
#pragma unroll
  for (int f = 0; f < 8; f++) O[f] = zero8;
  float mrow[8], lrow[8];  // lrow = per-lane PARTIAL sums until the end
#pragma unroll
  for (int r = 0; r < 8; r++) {
    mrow[r] = -1e30f;
    lrow[r] = 0.f;
  }

  const unsigned short* kb = kn + (size_t)cb * CTX * DIMX + h * HDIM;
  const unsigned short* vb = vT + (size_t)(cb * NHEADS + h) * HDIM * CTX;
  const float LOG2E = 1.44269504f;

  for (int l0 = 0; l0 < CTX; l0 += 32) {
    // scores 16 x 32 : two 16x16 C-frags, K=128 in 4 chunks
    v8f sc0 = zero8, sc1 = zero8;
#pragma unroll
    for (int kk = 0; kk < 4; kk++) {
      Frag16 b0, b1;
#pragma unroll
      for (int v = 0; v < 8; v++) {
        const int d0 = kk * 32 + hl * 16 + v * 2;
        b0.u[v] = *(const unsigned int*)(kb + (size_t)(l0 + lidx) * DIMX + d0);
        b1.u[v] = *(const unsigned int*)(kb + (size_t)(l0 + 16 + lidx) * DIMX + d0);
      }
      sc0 = __builtin_amdgcn_wmma_f32_16x16x32_bf16(false, qa[kk].v, false, b0.v,
                                                    (short)0, sc0, false, false);
      sc1 = __builtin_amdgcn_wmma_f32_16x16x32_bf16(false, qa[kk].v, false, b1.v,
                                                    (short)0, sc1, false, false);
    }

    // online softmax: only the row-MAX needs a cross-lane reduce per block
    float p0[8], p1[8], corr[8];
#pragma unroll
    for (int r = 0; r < 8; r++) {
      float bm = fmaxf(sc0[r], sc1[r]);
      bm = fmaxf(bm, __shfl_xor(bm, 1));
      bm = fmaxf(bm, __shfl_xor(bm, 2));
      bm = fmaxf(bm, __shfl_xor(bm, 4));
      bm = fmaxf(bm, __shfl_xor(bm, 8));
      const float mn = fmaxf(mrow[r], bm);
      corr[r] = exp2f((mrow[r] - mn) * LOG2E);
      p0[r]   = exp2f((sc0[r] - mn) * LOG2E);
      p1[r]   = exp2f((sc1[r] - mn) * LOG2E);
      lrow[r] = lrow[r] * corr[r] + (p0[r] + p1[r]);  // per-lane partial
      mrow[r] = mn;
    }
#pragma unroll
    for (int f = 0; f < 8; f++)
#pragma unroll
      for (int r = 0; r < 8; r++) O[f][r] *= corr[r];

    // P: C-layout -> LDS -> A-layout (bf16)
#pragma unroll
    for (int r = 0; r < 8; r++) {
      const int rr = r + hl * 8;
      Pt[rr * 34 + lidx]      = bf16one(p0[r]);
      Pt[rr * 34 + 16 + lidx] = bf16one(p1[r]);
    }
    __syncthreads();
    Frag16 pa;
#pragma unroll
    for (int v = 0; v < 8; v++) {
      const int k0 = hl * 16 + v * 2;
      pa.u[v] = *(const unsigned int*)&Pt[lidx * 34 + k0];
    }
    __syncthreads();  // WAR guard before next iteration's Pt writes

    // O += P(16x32) @ V(32x128): B-frags from per-head-transposed vT
#pragma unroll
    for (int f = 0; f < 8; f++) {
      Frag16 bv;
#pragma unroll
      for (int v = 0; v < 8; v++) {
        const int k0 = hl * 16 + v * 2;
        const int d  = f * 16 + lidx;
        bv.u[v] = *(const unsigned int*)(vb + (size_t)d * CTX + l0 + k0);
      }
      O[f] = __builtin_amdgcn_wmma_f32_16x16x32_bf16(false, pa.v, false, bv.v,
                                                     (short)0, O[f], false, false);
    }
  }

  // finish the deferred row-sum reduction (once, not per L-block)
#pragma unroll
  for (int r = 0; r < 8; r++) {
    float s = lrow[r];
    s += __shfl_xor(s, 1);
    s += __shfl_xor(s, 2);
    s += __shfl_xor(s, 4);
    s += __shfl_xor(s, 8);
    lrow[r] = s;
  }

  // normalize and write bf16 attention output [CB*S, D]
#pragma unroll
  for (int f = 0; f < 8; f++) {
#pragma unroll
    for (int r = 0; r < 8; r++) {
      const float o   = O[f][r] / lrow[r];
      const size_t gm = (size_t)(cb * SEQ + s0 + hl * 8 + r);
      const int gc    = h * HDIM + f * 16 + lidx;
      out[gm * DIMX + gc] = bf16one(o);
    }
  }
}

// ---------------------------------------------------------------------------
extern "C" void kernel_launch(void* const* d_in, const int* in_sizes, int n_in,
                              void* d_out, int out_size, void* d_ws,
                              size_t ws_size, hipStream_t stream) {
  const float* hid[2] = {(const float*)d_in[0], (const float*)d_in[1]};
  const float* ctx[2] = {(const float*)d_in[2], (const float*)d_in[3]};
  const float* Wq  = (const float*)d_in[4];
  const float* bq  = (const float*)d_in[5];
  const float* Wkv = (const float*)d_in[6];
  const float* bkv = (const float*)d_in[7];
  const float* gq  = (const float*)d_in[8];
  const float* gk  = (const float*)d_in[9];
  const float* Wo  = (const float*)d_in[10];
  const float* bo  = (const float*)d_in[11];
  float* outp = (float*)d_out;

  char* ws = (char*)d_ws;
  size_t off = 0;
  auto take = [&](size_t bytes) -> char* {
    char* p = ws + off;
    off = (off + bytes + 255) & ~(size_t)255;
    return p;
  };
  float* qf32  = (float*)take((size_t)CBATCH * SEQ * DIMX * 4);
  float* kvf32 = (float*)take((size_t)CBATCH * CTX * 2 * DIMX * 4);
  unsigned short* qn = (unsigned short*)take((size_t)CBATCH * SEQ * DIMX * 2);
  unsigned short* kn = (unsigned short*)take((size_t)CBATCH * CTX * DIMX * 2);
  unsigned short* vT =
      (unsigned short*)take((size_t)CBATCH * NHEADS * HDIM * CTX * 2);
  unsigned short* ao = (unsigned short*)take((size_t)CBATCH * SEQ * DIMX * 2);
  (void)ws_size; (void)in_sizes; (void)n_in; (void)out_size;

  // 1) projections (fp32 A -> bf16 WMMA -> fp32 out); grid.x = M-tiles so the
  //    L2-resident working set is the activation strip, weights stream once.
  for (int b = 0; b < 2; b++) {
    gemm_wmma_bf16<false><<<dim3(SEQ / 128, DIMX / 128), 256, 0, stream>>>(
        hid[b], Wq, bq, qf32 + (size_t)b * SEQ * DIMX, SEQ, DIMX, DIMX);
    gemm_wmma_bf16<false><<<dim3(CTX / 128, 2 * DIMX / 128), 256, 0, stream>>>(
        ctx[b], Wkv, bkv, kvf32 + (size_t)b * CTX * 2 * DIMX, CTX, 2 * DIMX, DIMX);
  }
  // 2) RMSNorm (Dh^-0.5 folded into q)
  const float qscale = 0.08838834764831845f;  // 128^-0.5
  rmsnorm_to_bf16<<<CBATCH * SEQ, 256, 0, stream>>>(qf32, DIMX, gq, qn, qscale);
  rmsnorm_to_bf16<<<CBATCH * CTX, 256, 0, stream>>>(kvf32, 2 * DIMX, gk, kn, 1.0f);
  // 3) per-head V transpose
  vtrans_kernel<<<dim3(CBATCH * CTX / 32, NHEADS), 256, 0, stream>>>(kvf32, vT);
  // 4) flash attention
  attn_kernel<<<dim3(SEQ / 16, NHEADS, CBATCH), 32, 0, stream>>>(qn, kn, vT, ao);
  // 5) output projection straight into d_out (cond/uncond halves contiguous)
  for (int b = 0; b < 2; b++) {
    gemm_wmma_bf16<true><<<dim3(SEQ / 128, DIMX / 128), 256, 0, stream>>>(
        ao + (size_t)b * SEQ * DIMX, Wo, bo, outp + (size_t)b * SEQ * DIMX, SEQ,
        DIMX, DIMX);
  }
}
